// CrossAttention_21947282883265
// MI455X (gfx1250) — matrix operand order repaired and checked by reading, not compiled, not run on hardware
//
#include <hip/hip_runtime.h>
#include <hip/hip_bf16.h>
#include <stdint.h>

typedef _Float16 v16h __attribute__((ext_vector_type(16)));
typedef _Float16 v8h  __attribute__((ext_vector_type(8)));
typedef float    v8f  __attribute__((ext_vector_type(8)));

#define NCAM 6
#define QTOT 1024
#define KCAM 1680
#define RQ   12288   // 2*6*1024 query rows
#define RK   20160   // 2*6*1680 key/value rows
#define ZR   2048    // 2*1024 BEV rows

union V16U { v16h v; v8h h[2]; };

__device__ __forceinline__ v8f wmma_f16(v16h a, v16h b, v8f c) {
  return __builtin_amdgcn_wmma_f32_16x16x32_f16(false, a, false, b, (short)0, c,
                                                false, false);
}

// A-matrix loader: 16x32 (MxK) f16, source row-major with row stride ld (halfs).
// ISA layout: lane%16 = row M; lanes 0-15 hold K 0-7 & 16-23, lanes 16-31 hold
// K 8-15 & 24-31 -> two contiguous 8-half chunks per lane.
__device__ __forceinline__ v16h load_A(const _Float16* base, int ld, int lane) {
  int row = lane & 15;
  int hi  = (lane >> 4) & 1;
  const _Float16* p = base + (size_t)row * ld + hi * 8;
  V16U u;
  u.h[0] = *(const v8h*)(p);
  u.h[1] = *(const v8h*)(p + 16);
  return u.v;
}

// B-matrix loader: 32x16 (KxN) f16, source stored "N-major": src[n][k], row
// stride ld. ISA layout: lane%16 = col N; lanes 0-15 hold K 0-7 & 16-23, lanes
// 16-31 hold K 8-15 & 24-31 -> two contiguous 8-half chunks per lane.
__device__ __forceinline__ v16h load_B(const _Float16* base, int ld, int lane) {
  int col = lane & 15;
  int hi  = (lane >> 4) & 1;
  const _Float16* p = base + (size_t)col * ld + hi * 8;
  V16U u;
  u.h[0] = *(const v8h*)(p);
  u.h[1] = *(const v8h*)(p + 16);
  return u.v;
}

// Async-stage a 32-row x 64-byte tile from global into LDS (per-wave private
// buffer). 128 b128 chunks = 32 lanes x 4 GLOBAL_LOAD_ASYNC_TO_LDS_B128.
// Tracked by ASYNCcnt; completes in order.
__device__ __forceinline__ void async_stage_tile(const _Float16* grow0,
                                                 size_t gstride_h,
                                                 _Float16* lrow0, int lane) {
#pragma unroll
  for (int j = 0; j < 4; ++j) {
    int c = j * 32 + lane;
    int r = c >> 2;              // tile row 0..31
    int off = (c & 3) * 8;       // halfs within the 64B row
    const _Float16* g = grow0 + (size_t)r * gstride_h + off;
    uint32_t l = (uint32_t)(uintptr_t)(lrow0 + r * 32 + off);
    asm volatile("global_load_async_to_lds_b128 %0, %1, off"
                 :: "v"(l), "v"(g) : "memory");
  }
}

__device__ __forceinline__ float redmax16(float v) {
  for (int m = 1; m <= 8; m <<= 1) v = fmaxf(v, __shfl_xor(v, m, 32));
  return v;
}
__device__ __forceinline__ float redsum16(float v) {
  for (int m = 1; m <= 8; m <<= 1) v += __shfl_xor(v, m, 32);
  return v;
}

// Transpose-cast f32 weight [K][N] -> f16 [N][K]
__global__ void wcast_kernel(const float* W, _Float16* Wt, int K, int N) {
  int t = blockIdx.x * blockDim.x + threadIdx.x;
  if (t >= K * N) return;
  int n = t / K, k = t % K;
  Wt[t] = (_Float16)W[(size_t)k * N + n];
}

// LayerNorm over 128 channels of a channel-major tensor [img][128][HW],
// writing row-major f16 [row][128]. One wave per row.
__global__ void ln_cast_kernel(const float* x, const float* g, const float* bt,
                               _Float16* y, int HW, int total_rows) {
  int wave = (blockIdx.x * blockDim.x + threadIdx.x) >> 5;
  int lane = threadIdx.x & 31;
  if (wave >= total_rows) return;
  int img = wave / HW, hw = wave % HW;
  const float* p = x + (size_t)img * 128 * HW + hw;
  float v[4];
  float s = 0.f;
  for (int j = 0; j < 4; ++j) { v[j] = p[(size_t)(lane + 32 * j) * HW]; s += v[j]; }
  for (int m = 16; m >= 1; m >>= 1) s += __shfl_xor(s, m, 32);
  float mu = s * (1.f / 128.f), vs = 0.f;
  for (int j = 0; j < 4; ++j) { float d = v[j] - mu; vs += d * d; }
  for (int m = 16; m >= 1; m >>= 1) vs += __shfl_xor(vs, m, 32);
  float rstd = rsqrtf(vs * (1.f / 128.f) + 1e-5f);
  _Float16* o = y + (size_t)wave * 128;
  for (int j = 0; j < 4; ++j) {
    int d = lane + 32 * j;
    o[d] = (_Float16)((v[j] - mu) * rstd * g[d] + bt[d]);
  }
}

// WMMA GEMM: C[MxN] = A[MxK] (f16 row-major) * Wt[NxK] (f16, transposed
// weights) + bias. One wave computes a 16x64 strip (4 accumulators): A tile
// loaded once per K-step, reused across 4 WMMAs.
// out_mode: 0 = f32 row-major, 1 = f16 row-major, 2 = f16 col-major ([N][M]).
__global__ void gemm_kernel(const _Float16* A, const _Float16* Wt,
                            const float* bias, void* out,
                            int M, int N, int K, int out_mode) {
  int wave = (blockIdx.x * blockDim.x + threadIdx.x) >> 5;
  int lane = threadIdx.x & 31;
  int ntiles = N >> 6;                  // 64-wide column strips
  int tiles = (M >> 4) * ntiles;
  if (wave >= tiles) return;
  int mt = wave / ntiles, nt = wave % ntiles;
  const _Float16* a_base = A + (size_t)mt * 16 * K;
  const _Float16* b_base = Wt + (size_t)nt * 64 * K;
  v8f acc[4] = {{}, {}, {}, {}};
  for (int k = 0; k < K; k += 32) {
    v16h a = load_A(a_base + k, K, lane);
#pragma unroll
    for (int j = 0; j < 4; ++j) {
      v16h b = load_B(b_base + (size_t)j * 16 * K + k, K, lane);
      acc[j] = wmma_f16(a, b, acc[j]);
    }
  }
  int rowbase = mt * 16 + ((lane >> 4) & 1) * 8;
#pragma unroll
  for (int j = 0; j < 4; ++j) {
    int col = nt * 64 + j * 16 + (lane & 15);
    float bsv = bias ? bias[col] : 0.f;
    if (out_mode == 0) {
      float* O = (float*)out;
      for (int i = 0; i < 8; ++i)
        O[(size_t)(rowbase + i) * N + col] = acc[j][i] + bsv;
    } else if (out_mode == 1) {
      _Float16* O = (_Float16*)out;
      for (int i = 0; i < 8; ++i)
        O[(size_t)(rowbase + i) * N + col] = (_Float16)(acc[j][i] + bsv);
    } else {
      _Float16* O = (_Float16*)out;
      for (int i = 0; i < 8; ++i)
        O[(size_t)col * M + rowbase + i] = (_Float16)(acc[j][i] + bsv);
    }
  }
}

// Flash-attention. grid = (16, 8): x -> query tile group, y -> (b*4+head).
// 4 waves/block, each wave owns a 16-query tile and streams all 6*1680 keys.
// K/V 32-key tiles are double-buffered in per-wave LDS via async-to-LDS loads.
__global__ void attn_kernel(const _Float16* qh, const _Float16* kh,
                            const _Float16* vt, _Float16* aout) {
  __shared__ __align__(16) _Float16 plds[4][16][32];      // P transpose
  __shared__ __align__(16) _Float16 kbuf[4][2][32][32];   // [key][dim]
  __shared__ __align__(16) _Float16 vbuf[4][2][32][32];   // [dim][key]
  int w    = threadIdx.x >> 5;
  int lane = threadIdx.x & 31;
  int hi   = lane >> 4;
  int col  = lane & 15;
  int bm = blockIdx.y;
  int b = bm >> 2, m = bm & 3;
  int qbase = (blockIdx.x * 4 + w) * 16;
  const int Mv = RK;                       // vt is [128][RK] col-major
  v8f acc0 = {}, acc1 = {};
  float mrun[8], lrun[8];
  for (int i = 0; i < 8; ++i) { mrun[i] = -1e30f; lrun[i] = 0.f; }
  const float scale = 0.17677669529663687f;  // 1/sqrt(32)
  const int nblk = (KCAM + 31) / 32;         // 53 (last block = 16 keys)

  for (int n = 0; n < NCAM; ++n) {
    const _Float16* qb =
        qh + ((size_t)((b * NCAM + n) * QTOT + qbase)) * 128 + m * 32;
    v16h aq = load_A(qb, 128, lane);
    size_t krow0 = (size_t)(b * NCAM + n) * KCAM;
    const _Float16* vcol0 = vt + ((size_t)m * 32) * Mv + krow0;
    // preload block 0 (4+4 async b128 instructions)
    async_stage_tile(kh + krow0 * 128 + m * 32, 128, &kbuf[w][0][0][0], lane);
    async_stage_tile(vcol0, Mv, &vbuf[w][0][0][0], lane);
    for (int ib = 0; ib < nblk; ++ib) {
      int kb = ib * 32;
      int pb = ib & 1;
      bool more = (ib + 1) < nblk;
      if (more) {
        // prefetch block ib+1 into the other buffer, then wait until only
        // those 8 newest async ops remain outstanding (in-order completion
        // => block ib fully resident in LDS).
        async_stage_tile(kh + (krow0 + kb + 32) * 128 + m * 32, 128,
                         &kbuf[w][pb ^ 1][0][0], lane);
        async_stage_tile(vcol0 + kb + 32, Mv, &vbuf[w][pb ^ 1][0][0], lane);
        asm volatile("s_wait_asynccnt 0x8" ::: "memory");
      } else {
        asm volatile("s_wait_asynccnt 0x0" ::: "memory");
      }
      bool full = (kb + 32) <= KCAM;
      const _Float16* kl = &kbuf[w][pb][0][0];
      v8f zero = {};
      v16h bk0 = load_B(kl, 32, lane);               // keys kb..kb+15
      v8f s0 = wmma_f16(aq, bk0, zero);
      v8f s1 = {};
      if (full) {
        v16h bk1 = load_B(kl + 16 * 32, 32, lane);   // keys kb+16..kb+31
        s1 = wmma_f16(aq, bk1, zero);
      }
      float p0[8], p1[8];
      for (int i = 0; i < 8; ++i) {
        float a0 = s0[i] * scale;
        float a1 = full ? s1[i] * scale : -1e30f;
        float tm = redmax16(fmaxf(a0, a1));
        float mn = fmaxf(mrun[i], tm);
        float fac = __expf(mrun[i] - mn);
        float e0 = __expf(a0 - mn);
        float e1 = full ? __expf(a1 - mn) : 0.f;
        float rs = redsum16(e0 + e1);
        lrun[i] = lrun[i] * fac + rs;
        mrun[i] = mn;
        acc0[i] *= fac;
        acc1[i] *= fac;
        p0[i] = e0;
        p1[i] = e1;
      }
      // C-layout -> A-layout transpose of P through wave-private LDS tile
      for (int i = 0; i < 8; ++i) {
        plds[w][hi * 8 + i][col]      = (_Float16)p0[i];
        plds[w][hi * 8 + i][16 + col] = (_Float16)p1[i];
      }
      asm volatile("s_wait_dscnt 0" ::: "memory");
      V16U u;
      const _Float16* pr = &plds[w][lane & 15][0];
      u.h[0] = *(const v8h*)(pr + hi * 8);
      u.h[1] = *(const v8h*)(pr + 16 + hi * 8);
      v16h ap = u.v;
      const _Float16* vl = &vbuf[w][pb][0][0];
      v16h bv0 = load_B(vl, 32, lane);               // dh 0..15
      v16h bv1 = load_B(vl + 16 * 32, 32, lane);     // dh 16..31
      acc0 = wmma_f16(ap, bv0, acc0);
      acc1 = wmma_f16(ap, bv1, acc1);
    }
  }
  for (int i = 0; i < 8; ++i) {
    int row = b * QTOT + qbase + hi * 8 + i;
    float inv = 1.f / lrun[i];
    aout[(size_t)row * 128 + m * 32 + col]      = (_Float16)(acc0[i] * inv);
    aout[(size_t)row * 128 + m * 32 + 16 + col] = (_Float16)(acc1[i] * inv);
  }
}

// z = LN(z0 + skip) -> f32 + f16 copies. One wave per row.
__global__ void pre_stage_kernel(const float* z0, const float* skip,
                                 const float* g, const float* bt,
                                 float* zpre, _Float16* zpreh) {
  int wave = (blockIdx.x * blockDim.x + threadIdx.x) >> 5;
  int lane = threadIdx.x & 31;
  if (wave >= ZR) return;
  int b = wave >> 10, hw = wave & 1023;
  float v[4];
  float s = 0.f;
  for (int j = 0; j < 4; ++j) {
    int d = lane + 32 * j;
    v[j] = z0[(size_t)wave * 128 + d] + skip[((size_t)(b * 128 + d)) * 1024 + hw];
    s += v[j];
  }
  for (int m = 16; m >= 1; m >>= 1) s += __shfl_xor(s, m, 32);
  float mu = s * (1.f / 128.f), vs = 0.f;
  for (int j = 0; j < 4; ++j) { float d = v[j] - mu; vs += d * d; }
  for (int m = 16; m >= 1; m >>= 1) vs += __shfl_xor(vs, m, 32);
  float rstd = rsqrtf(vs * (1.f / 128.f) + 1e-5f);
  for (int j = 0; j < 4; ++j) {
    int d = lane + 32 * j;
    float y = (v[j] - mu) * rstd * g[d] + bt[d];
    zpre[(size_t)wave * 128 + d]  = y;
    zpreh[(size_t)wave * 128 + d] = (_Float16)y;
  }
}

__global__ void gelu_kernel(const float* h, _Float16* o, int n) {
  int t = blockIdx.x * blockDim.x + threadIdx.x;
  if (t >= n) return;
  float x = h[t];
  o[t] = (_Float16)(0.5f * x * (1.f + erff(x * 0.70710678118654752f)));
}

// out[(b,d,H,W)] = postLN(zpre + mlp). One wave per row, transposed store.
__global__ void final_kernel(const float* zpre, const float* mlp,
                             const float* g, const float* bt, float* out) {
  int wave = (blockIdx.x * blockDim.x + threadIdx.x) >> 5;
  int lane = threadIdx.x & 31;
  if (wave >= ZR) return;
  int b = wave >> 10, hw = wave & 1023;
  float v[4];
  float s = 0.f;
  for (int j = 0; j < 4; ++j) {
    int d = lane + 32 * j;
    v[j] = zpre[(size_t)wave * 128 + d] + mlp[(size_t)wave * 128 + d];
    s += v[j];
  }
  for (int m = 16; m >= 1; m >>= 1) s += __shfl_xor(s, m, 32);
  float mu = s * (1.f / 128.f), vs = 0.f;
  for (int j = 0; j < 4; ++j) { float d = v[j] - mu; vs += d * d; }
  for (int m = 16; m >= 1; m >>= 1) vs += __shfl_xor(vs, m, 32);
  float rstd = rsqrtf(vs * (1.f / 128.f) + 1e-5f);
  for (int j = 0; j < 4; ++j) {
    int d = lane + 32 * j;
    out[((size_t)(b * 128 + d)) * 1024 + hw] = (v[j] - mu) * rstd * g[d] + bt[d];
  }
}

extern "C" void kernel_launch(void* const* d_in, const int* in_sizes, int n_in,
                              void* d_out, int out_size, void* d_ws,
                              size_t ws_size, hipStream_t stream) {
  const float* q    = (const float*)d_in[0];
  const float* k    = (const float*)d_in[1];
  const float* v    = (const float*)d_in[2];
  const float* skip = (const float*)d_in[3];
  const float* qg   = (const float*)d_in[4];
  const float* qbt  = (const float*)d_in[5];
  const float* Wq   = (const float*)d_in[6];
  const float* bq   = (const float*)d_in[7];
  const float* kg   = (const float*)d_in[8];
  const float* kbt  = (const float*)d_in[9];
  const float* Wk   = (const float*)d_in[10];
  const float* bk   = (const float*)d_in[11];
  const float* vg   = (const float*)d_in[12];
  const float* vbt  = (const float*)d_in[13];
  const float* Wv   = (const float*)d_in[14];
  const float* bvv  = (const float*)d_in[15];
  const float* Wp   = (const float*)d_in[16];
  const float* bp   = (const float*)d_in[17];
  const float* preg = (const float*)d_in[18];
  const float* preb = (const float*)d_in[19];
  const float* W1   = (const float*)d_in[20];
  const float* b1   = (const float*)d_in[21];
  const float* W2   = (const float*)d_in[22];
  const float* b2   = (const float*)d_in[23];
  const float* postg= (const float*)d_in[24];
  const float* postb= (const float*)d_in[25];
  float* out = (float*)d_out;

  uint8_t* p = (uint8_t*)d_ws;
  auto carve = [&](size_t bytes) -> void* {
    void* r = (void*)p;
    p += (bytes + 255) & ~(size_t)255;
    return r;
  };
  _Float16* Wq_t  = (_Float16*)carve(128 * 128 * 2);
  _Float16* Wk_t  = (_Float16*)carve(128 * 128 * 2);
  _Float16* Wv_t  = (_Float16*)carve(128 * 128 * 2);
  _Float16* Wp_t  = (_Float16*)carve(128 * 128 * 2);
  _Float16* W1_t  = (_Float16*)carve(128 * 256 * 2);
  _Float16* W2_t  = (_Float16*)carve(256 * 128 * 2);
  _Float16* q_ln  = (_Float16*)carve((size_t)RQ * 128 * 2);
  _Float16* k_ln  = (_Float16*)carve((size_t)RK * 128 * 2);
  _Float16* v_ln  = (_Float16*)carve((size_t)RK * 128 * 2);
  _Float16* qh    = (_Float16*)carve((size_t)RQ * 128 * 2);
  _Float16* khb   = (_Float16*)carve((size_t)RK * 128 * 2 + 4096); // +tail slack
  _Float16* vtb   = (_Float16*)carve((size_t)128 * RK * 2 + 4096); // +tail slack
  _Float16* ah    = (_Float16*)carve((size_t)ZR * 128 * 2);
  float*    z0    = (float*)carve((size_t)ZR * 128 * 4);
  float*    zpre  = (float*)carve((size_t)ZR * 128 * 4);
  _Float16* zpreh = (_Float16*)carve((size_t)ZR * 128 * 2);
  float*    hbuf  = (float*)carve((size_t)ZR * 256 * 4);
  _Float16* hg    = (_Float16*)carve((size_t)ZR * 256 * 2);
  float*    mlp   = (float*)carve((size_t)ZR * 128 * 4);

  // 1) weight transposes (f32 -> f16 [N][K])
  wcast_kernel<<<(128 * 128 + 255) / 256, 256, 0, stream>>>(Wq, Wq_t, 128, 128);
  wcast_kernel<<<(128 * 128 + 255) / 256, 256, 0, stream>>>(Wk, Wk_t, 128, 128);
  wcast_kernel<<<(128 * 128 + 255) / 256, 256, 0, stream>>>(Wv, Wv_t, 128, 128);
  wcast_kernel<<<(128 * 128 + 255) / 256, 256, 0, stream>>>(Wp, Wp_t, 128, 128);
  wcast_kernel<<<(128 * 256 + 255) / 256, 256, 0, stream>>>(W1, W1_t, 128, 256);
  wcast_kernel<<<(256 * 128 + 255) / 256, 256, 0, stream>>>(W2, W2_t, 256, 128);

  // 2) fused LayerNorm + transpose-gather + f16 cast
  ln_cast_kernel<<<(RQ + 7) / 8, 256, 0, stream>>>(q, qg, qbt, q_ln, QTOT, RQ);
  ln_cast_kernel<<<(RK + 7) / 8, 256, 0, stream>>>(k, kg, kbt, k_ln, KCAM, RK);
  ln_cast_kernel<<<(RK + 7) / 8, 256, 0, stream>>>(v, vg, vbt, v_ln, KCAM, RK);

  auto gblocks = [](int M, int N) { return ((M / 16) * (N / 64) + 7) / 8; };
  // 3) QKV projections (WMMA)
  gemm_kernel<<<gblocks(RQ, 128), 256, 0, stream>>>(q_ln, Wq_t, bq, qh,
                                                    RQ, 128, 128, 1);
  gemm_kernel<<<gblocks(RK, 128), 256, 0, stream>>>(k_ln, Wk_t, bk, khb,
                                                    RK, 128, 128, 1);
  gemm_kernel<<<gblocks(RK, 128), 256, 0, stream>>>(v_ln, Wv_t, bvv, vtb,
                                                    RK, 128, 128, 2);
  // 4) flash attention (WMMA QK^T + PV, online softmax, async K/V staging)
  attn_kernel<<<dim3(16, 8), 128, 0, stream>>>(qh, khb, vtb, ah);
  // 5) output projection
  gemm_kernel<<<gblocks(ZR, 128), 256, 0, stream>>>(ah, Wp_t, bp, z0,
                                                    ZR, 128, 128, 0);
  // 6) +skip, pre-LN
  pre_stage_kernel<<<ZR / 8, 256, 0, stream>>>(z0, skip, preg, preb, zpre, zpreh);
  // 7) MLP
  gemm_kernel<<<gblocks(ZR, 256), 256, 0, stream>>>(zpreh, W1_t, b1, hbuf,
                                                    ZR, 256, 128, 0);
  gelu_kernel<<<(ZR * 256 + 255) / 256, 256, 0, stream>>>(hbuf, hg, ZR * 256);
  gemm_kernel<<<gblocks(ZR, 128), 256, 0, stream>>>(hg, W2_t, b2, mlp,
                                                    ZR, 128, 256, 0);
  // 8) residual + post-LN + transposed store
  final_kernel<<<ZR / 8, 256, 0, stream>>>(zpre, mlp, postg, postb, out);
}
